// MaskedOmics_49297634623952
// MI455X (gfx1250) — matrix operand order repaired and checked by reading, not compiled, not run on hardware
//
#include <hip/hip_runtime.h>
#include <hip/hip_bf16.h>

typedef __attribute__((ext_vector_type(16))) _Float16 v16h;
typedef __attribute__((ext_vector_type(8)))  _Float16 v8h;
typedef __attribute__((ext_vector_type(8)))  float    v8f;
typedef __attribute__((__vector_size__(16))) int      v4i;

// Problem dimensions (fixed by the reference)
constexpr int Bn  = 1024;   // batch
constexpr int IN  = 4096;   // input dim
constexpr int Pn  = 512;    // pathways
constexpr int D1  = 8;
constexpr int D2  = 16;
constexpr int H1  = D1 * Pn;   // 4096
constexpr int H2  = D2 * Pn;   // 8192
constexpr int H3  = H2 / 4;    // 2048
constexpr int Cn  = 4;         // classes

// ---------------------------------------------------------------------------
// CDNA5 async global->LDS copy (ASYNCcnt) with sync fallback.
// Builtin signature (from hipcc diagnostic): param0 = AS(1) v4i* (global src),
// then LDS dst, imm offset, imm cpol.
// ---------------------------------------------------------------------------
#if __has_builtin(__builtin_amdgcn_global_load_async_to_lds_b128)
#define HAS_ASYNC_LDS 1
#else
#define HAS_ASYNC_LDS 0
#endif

typedef __attribute__((address_space(1))) v4i as1_v4i;
typedef __attribute__((address_space(3))) v4i as3_v4i;

__device__ __forceinline__ void async_copy16(const _Float16* __restrict__ g,
                                             _Float16* l) {
#if HAS_ASYNC_LDS
    __builtin_amdgcn_global_load_async_to_lds_b128(
        (as1_v4i*)(uintptr_t)g,
        (as3_v4i*)(uint32_t)(uintptr_t)l,
        /*offset=*/0, /*cpol=*/0);
#else
    *(uint4*)l = *(const uint4*)g;
#endif
}

template <int N>
__device__ __forceinline__ void wait_async() {
#if HAS_ASYNC_LDS
#if __has_builtin(__builtin_amdgcn_s_wait_asynccnt)
    __builtin_amdgcn_s_wait_asynccnt(N);
#else
    asm volatile("s_wait_asynccnt %0" ::"n"(N) : "memory");
#endif
#endif
}

// ---------------------------------------------------------------------------
// Transpose + f32->f16 convert, optional pathway mask fused (for fc1_w).
// src: [R, Cc] f32 row-major ->  dst: [Cc, R] f16 row-major  (dst[c][r]=src[r][c])
// mask (if comp != nullptr): zero element when comp[r*P + (c>>log2D)] == 0
// ---------------------------------------------------------------------------
__global__ __launch_bounds__(256) void pack_transpose(
    const float* __restrict__ src, _Float16* __restrict__ dst,
    int R, int Cc, const int* __restrict__ comp, int P, int log2D)
{
    __shared__ _Float16 tile[32][33];
    const int c0 = blockIdx.x * 32;
    const int r0 = blockIdx.y * 32;
    const int tx = threadIdx.x;     // 0..31
    const int ty = threadIdx.y;     // 0..7
#pragma unroll
    for (int rr = 0; rr < 4; ++rr) {
        const int rl = ty * 4 + rr;
        const int r  = r0 + rl;
        const int c  = c0 + tx;
        float v = src[(size_t)r * Cc + c];
        if (comp) {
            if (comp[(size_t)r * P + (c >> log2D)] == 0) v = 0.0f;
        }
        tile[rl][tx] = (_Float16)v;
    }
    __syncthreads();
#pragma unroll
    for (int rr = 0; rr < 4; ++rr) {
        const int cl = ty * 4 + rr;                 // local column
        dst[(size_t)(c0 + cl) * R + (r0 + tx)] = tile[tx][cl];
    }
}

// ---------------------------------------------------------------------------
// Elementwise f32 -> f16 (vectorized by 4)
// ---------------------------------------------------------------------------
__global__ __launch_bounds__(256) void convert_f16(
    const float* __restrict__ src, _Float16* __restrict__ dst, int n4)
{
    const int i = blockIdx.x * blockDim.x + threadIdx.x;
    if (i < n4) {
        const float4 v = ((const float4*)src)[i];
        dst[i * 4 + 0] = (_Float16)v.x;
        dst[i * 4 + 1] = (_Float16)v.y;
        dst[i * 4 + 2] = (_Float16)v.z;
        dst[i * 4 + 3] = (_Float16)v.w;
    }
}

// ---------------------------------------------------------------------------
// WMMA GEMM with double-buffered async LDS staging:
//   C[M,N] = A[M,K] @ Bt[N,K]^T + bias[N]   (f16 in, f32 accumulate)
// Block: 256 threads = 8 waves (2 M-waves x 4 N-waves). Block tile 64x256, K-step 32.
// LDS: sA[2][64][32] (8KB) + sB[2][256][32] (32KB); async copies tracked by ASYNCcnt.
// Fragment layouts per CDNA5 ISA 7.12.2:
//   A (16x32): lane<16 -> M=lane, halves = K{k0..k0+7, k0+16..k0+23}
//              lane>=16 -> M=lane-16, halves = K{k0+8..k0+15, k0+24..k0+31}
//   B (32x16): lane<16 -> N=lane, K as A lane<16 ; lane>=16 -> N=lane-16, K as A lane>=16
//   C/D: lane<16 -> rows 0..7 col lane ; lane>=16 -> rows 8..15 col lane-16
// ---------------------------------------------------------------------------
template <bool RELU, bool OUTF16>
__global__ __launch_bounds__(256) void gemm_wmma(
    const _Float16* __restrict__ A,   // [M, K]
    const _Float16* __restrict__ Bt,  // [N, K]  (== B^T)
    const float*    __restrict__ bias,// [N]
    float*          __restrict__ Cf32,
    _Float16*       __restrict__ Cf16,
    int M, int N, int K)
{
    __shared__ alignas(16) _Float16 sA[2][64][32];
    __shared__ alignas(16) _Float16 sB[2][256][32];

    const int t     = threadIdx.x;
    const int lane  = t & 31;
    const int wave  = t >> 5;
    const int waveM = wave & 1;    // 0..1
    const int waveN = wave >> 1;   // 0..3
    const int lm    = lane & 15;
    const int hi    = lane >> 4;   // 0 or 1

    const int mBlock = blockIdx.y * 64;
    const int nBlock = blockIdx.x * 256;

    // Per-thread copy assignment: 1 x 16B chunk of the A tile, 4 of the B tile.
    const int cRow = t >> 2;       // 0..63
    const int cCh  = (t & 3) * 8;  // half-element offset within the 32-wide K slab

    auto copy_tiles = [&](int buf, int k0) {
        async_copy16(A + (size_t)(mBlock + cRow) * K + k0 + cCh,
                     &sA[buf][cRow][cCh]);
#pragma unroll
        for (int j = 0; j < 4; ++j) {
            const int r = cRow + j * 64;   // 0..255
            async_copy16(Bt + (size_t)(nBlock + r) * K + k0 + cCh,
                         &sB[buf][r][cCh]);
        }
    };

    v8f acc[2][4] = {};

    const int nk = K / 32;
    copy_tiles(0, 0);                      // preload buffer 0 (5 async ops)

    for (int kt = 0; kt < nk; ++kt) {
        const int buf = kt & 1;
        if (kt + 1 < nk) {
            copy_tiles(buf ^ 1, (kt + 1) * 32);  // 5 more in flight
            wait_async<5>();   // oldest 5 (current buf) retired; newest 5 pending
        } else {
            wait_async<0>();
        }
        __syncthreads();       // publish LDS writes to all waves

        v16h a[2];
#pragma unroll
        for (int mi = 0; mi < 2; ++mi) {
            const int ml = waveM * 32 + mi * 16 + lm;
            const v8h lo = *(const v8h*)&sA[buf][ml][hi * 8];
            const v8h hh = *(const v8h*)&sA[buf][ml][16 + hi * 8];
            a[mi] = __builtin_shufflevector(lo, hh, 0, 1, 2, 3, 4, 5, 6, 7,
                                            8, 9, 10, 11, 12, 13, 14, 15);
        }
        v16h b[4];
#pragma unroll
        for (int ni = 0; ni < 4; ++ni) {
            const int nl = waveN * 64 + ni * 16 + lm;
            const v8h lo = *(const v8h*)&sB[buf][nl][hi * 8];
            const v8h hh = *(const v8h*)&sB[buf][nl][16 + hi * 8];
            b[ni] = __builtin_shufflevector(lo, hh, 0, 1, 2, 3, 4, 5, 6, 7,
                                            8, 9, 10, 11, 12, 13, 14, 15);
        }
#pragma unroll
        for (int mi = 0; mi < 2; ++mi)
#pragma unroll
            for (int ni = 0; ni < 4; ++ni)
                acc[mi][ni] = __builtin_amdgcn_wmma_f32_16x16x32_f16(
                    /*neg_a=*/false, a[mi], /*neg_b=*/false, b[ni],
                    /*c_mod=*/(short)0, acc[mi][ni],
                    /*reuse_a=*/false, /*reuse_b=*/false);

        __syncthreads();       // all reads of buf done before it is overwritten
    }

    // Epilogue: bias (+ReLU), scatter per C/D layout.
#pragma unroll
    for (int mi = 0; mi < 2; ++mi) {
#pragma unroll
        for (int ni = 0; ni < 4; ++ni) {
            const int n  = nBlock + waveN * 64 + ni * 16 + lm;
            const float bv = bias[n];
            const int rBase = mBlock + waveM * 32 + mi * 16 + hi * 8;
#pragma unroll
            for (int r = 0; r < 8; ++r) {
                float v = acc[mi][ni][r] + bv;
                if (RELU) v = v > 0.0f ? v : 0.0f;
                const size_t idx = (size_t)(rBase + r) * N + n;
                if (OUTF16) Cf16[idx] = (_Float16)v;
                else        Cf32[idx] = v;
            }
        }
    }
}

// ---------------------------------------------------------------------------
// Block-diagonal GEMM2: out2[m,n] = fc2_b[n] + sum_{i<8} out1[m, p*8+i] * fc2_w[p*8+i, n]
// where p = n / 16.  (mask_2 = kron(I_P, ones(8,16)))  Output f16 (feeds GEMM3 A).
// ---------------------------------------------------------------------------
__global__ __launch_bounds__(256) void gemm2_blockdiag(
    const float* __restrict__ out1,   // [B, H1] f32
    const float* __restrict__ fc2_w,  // [H1, H2] f32
    const float* __restrict__ fc2_b,  // [H2]
    _Float16*    __restrict__ out2h)  // [B, H2] f16
{
    const int idx = blockIdx.x * 256 + threadIdx.x;   // over B*H2
    const int n = idx & (H2 - 1);
    const int m = idx >> 13;                          // /H2
    const int p = n >> 4;                             // D2 = 16
    float acc = fc2_b[n];
    const float* x = out1 + (size_t)m * H1 + p * D1;
    const float* w = fc2_w + (size_t)(p * D1) * H2 + n;
#pragma unroll
    for (int i = 0; i < D1; ++i)
        acc += x[i] * w[(size_t)i * H2];
    out2h[idx] = (_Float16)acc;
}

// ---------------------------------------------------------------------------
// Final: logits[m, c] = b4[c] + sum_k h[m,k] * w4[k,c]   (H3=2048, C=4)
// One block per row m; 256 threads x 8 contiguous k each; LDS tree reduction.
// ---------------------------------------------------------------------------
__global__ __launch_bounds__(256) void final_logits(
    const _Float16* __restrict__ h,   // [B, H3] f16
    const float*    __restrict__ w4,  // [H3, C] f32
    const float*    __restrict__ b4,  // [C]
    float*          __restrict__ out) // [B, C]
{
    __shared__ float red[256][Cn];
    const int m = blockIdx.x;
    const int t = threadIdx.x;
    float acc[Cn] = {0.f, 0.f, 0.f, 0.f};
    const _Float16* hp = h + (size_t)m * H3 + t * 8;
    const float*    wp = w4 + (size_t)t * 8 * Cn;
#pragma unroll
    for (int j = 0; j < 8; ++j) {
        const float hv = (float)hp[j];
#pragma unroll
        for (int c = 0; c < Cn; ++c) acc[c] += hv * wp[j * Cn + c];
    }
#pragma unroll
    for (int c = 0; c < Cn; ++c) red[t][c] = acc[c];
    __syncthreads();
    for (int s = 128; s > 0; s >>= 1) {
        if (t < s) {
#pragma unroll
            for (int c = 0; c < Cn; ++c) red[t][c] += red[t + s][c];
        }
        __syncthreads();
    }
    if (t < Cn) out[(size_t)m * Cn + t] = red[0][t] + b4[t];
}

// ---------------------------------------------------------------------------
// Launch: workspace layout (bytes)
//   [0,32M)    W1t  f16 [H1, IN]   (masked fc1_w, transposed)
//   [32M,40M)  Xh   f16 [B, IN]
//   [40M,56M)  out1 f32 [B, H1]
//   [56M,72M)  out2 f16 [B, H2]
//   [72M,104M) W3t  f16 [H3, H2]
//   [104M,108M) h   f16 [B, H3]
// ---------------------------------------------------------------------------
extern "C" void kernel_launch(void* const* d_in, const int* in_sizes, int n_in,
                              void* d_out, int out_size, void* d_ws, size_t ws_size,
                              hipStream_t stream) {
    const float* data  = (const float*)d_in[0];
    const int*   comp  = (const int*)  d_in[1];
    const float* fc1_w = (const float*)d_in[2];
    const float* fc1_b = (const float*)d_in[3];
    const float* fc2_w = (const float*)d_in[4];
    const float* fc2_b = (const float*)d_in[5];
    const float* w3    = (const float*)d_in[6];
    const float* b3    = (const float*)d_in[7];
    const float* w4    = (const float*)d_in[8];
    const float* b4    = (const float*)d_in[9];
    float* logits = (float*)d_out;

    char* ws = (char*)d_ws;
    _Float16* W1t  = (_Float16*)(ws + 0);
    _Float16* Xh   = (_Float16*)(ws + (32ull << 20));
    float*    out1 = (float*)   (ws + (40ull << 20));
    _Float16* out2 = (_Float16*)(ws + (56ull << 20));
    _Float16* W3t  = (_Float16*)(ws + (72ull << 20));
    _Float16* h16  = (_Float16*)(ws + (104ull << 20));

    // 1) Pack weights (transpose + f16 convert; mask fused for fc1_w)
    pack_transpose<<<dim3(H1 / 32, IN / 32), dim3(32, 8), 0, stream>>>(
        fc1_w, W1t, IN, H1, comp, Pn, 3 /* log2(D1) */);
    pack_transpose<<<dim3(H3 / 32, H2 / 32), dim3(32, 8), 0, stream>>>(
        w3, W3t, H2, H3, nullptr, 0, 0);

    // 2) Activations to f16
    convert_f16<<<(Bn * IN / 4 + 255) / 256, 256, 0, stream>>>(data, Xh, Bn * IN / 4);

    // 3) GEMM1: out1 = Xh @ W1t^T + fc1_b   (f32 out)
    gemm_wmma<false, false><<<dim3(H1 / 256, Bn / 64), 256, 0, stream>>>(
        Xh, W1t, fc1_b, out1, nullptr, Bn, H1, IN);

    // 4) Block-diagonal GEMM2 (+bias) -> f16
    gemm2_blockdiag<<<(Bn * H2) / 256, 256, 0, stream>>>(out1, fc2_w, fc2_b, out2);

    // 5) GEMM3: h = relu(out2 @ W3t^T + b3)  (f16 out)
    gemm_wmma<true, true><<<dim3(H3 / 256, Bn / 64), 256, 0, stream>>>(
        out2, W3t, b3, nullptr, h16, Bn, H3, H2);

    // 6) logits = h @ w4 + b4
    final_logits<<<Bn, 256, 0, stream>>>(h16, w4, b4, logits);
}
